// DSSConf_22230750724541
// MI455X (gfx1250) — compile-verified, matrix operand order checked by reading, not run on hardware
//
#include <hip/hip_runtime.h>
#include <hip/hip_bf16.h>
#include <math.h>

// ---------------- problem constants (match reference) ----------------
#define HIDDEN 256
#define NG     50
#define NGP    64        // NG padded to multiple of 32 for WMMA K
#define NF     128
#define NCONF  100000
#define NGRAPH 10000
#define ECONF  1000000
#define EGRAPH 30000
#define PER_G  (NCONF / NGRAPH)   // 10
#define PI_F   3.14159265358979f
#define CUTOFF 10.0f

typedef __bf16 bf16_t;
typedef __bf16 v16bf __attribute__((ext_vector_type(16)));
typedef __bf16 v8bf  __attribute__((ext_vector_type(8)));
typedef float  v8f   __attribute__((ext_vector_type(8)));
typedef float  v4f   __attribute__((ext_vector_type(4)));
typedef unsigned int u32x4 __attribute__((ext_vector_type(4)));
typedef int          i32x4 __attribute__((ext_vector_type(4)));
typedef int          i32x8 __attribute__((ext_vector_type(8)));

// ---------------- WMMA helpers (CDNA5 16x16x32 bf16, f32 accum) ----------------
// 16-bit A-matrix 16x32 layout (ISA 7.12.2): lanes 0-15 hold rows M=0..15 with
// K runs [0..7],[16..23]; lanes 16-31 hold K runs [8..15],[24..31]. B (stored
// N-major / transposed in memory) uses the same per-lane run structure with the
// row index = N. Each fragment therefore is two contiguous 16-byte loads.
__device__ __forceinline__ v16bf load_frag(const bf16_t* base, int ld, int k0) {
    int lane = threadIdx.x & 31;
    int half = lane >> 4;        // 0: K runs {0,16}, 1: K runs {8,24}
    int r    = lane & 15;        // M (or N) index
    const bf16_t* p = base + r * ld + k0;
    v8bf lo = *(const v8bf*)(p + (half ? 8  : 0));
    v8bf hi = *(const v8bf*)(p + (half ? 24 : 16));
    v16bf f;
#pragma unroll
    for (int i = 0; i < 8; ++i) { f[i] = lo[i]; f[i + 8] = hi[i]; }
    return f;
}

__device__ __forceinline__ v8f wmma32(v16bf a, v16bf b, v8f c) {
    return __builtin_amdgcn_wmma_f32_16x16x32_bf16(
        /*neg_a=*/false, a, /*neg_b=*/false, b,
        /*c_mod=*/(short)0, c, /*reuse_a=*/false, /*reuse_b=*/false);
}

// ---------------- Tensor Data Mover: 2-D fp32 tile -> LDS --------------------
// Builds a D# per CDNA5 ISA 8.3/8.4: group0 = {count=1, lds_addr, global_addr,
// type=2}; group1 = {data_size=4B, tensor dims, tile dims, dim0 stride};
// groups 2/3 zero (2-D tensor). Issued by one wave; tracked with TENSORcnt.
// This toolchain exposes the 6-argument builtin (g0, g1, g2, g3, g4, cpol).
__device__ __forceinline__ void tdm_load_tile_f32(const void* gtile, void* ldsdst,
                                                  unsigned tensor_d0, unsigned tensor_d1,
                                                  unsigned tile_d0, unsigned tile_d1,
                                                  unsigned stride0) {
    unsigned long long ga = (unsigned long long)(uintptr_t)gtile;
    unsigned lds = (unsigned)(uintptr_t)ldsdst;     // low 32 bits = LDS byte addr
    u32x4 g0;
    g0[0] = 1u;                                     // count=1, user mode, no gather
    g0[1] = lds;                                    // lds_addr  (bits 63:32)
    g0[2] = (unsigned)(ga & 0xffffffffu);           // global_addr[31:0]
    g0[3] = (unsigned)((ga >> 32) & 0x01ffffffu)    // global_addr[56:32]
          | (2u << 30);                             // type=2 ("image")
    i32x8 g1;
    g1[0] = (int)(2u << 16);                        // wg_mask=0, data_size=2 (4B)
    g1[1] = (int)((tensor_d0 & 0xffffu) << 16);     // tensor_dim0[15:0] @ bits63:48
    g1[2] = (int)((tensor_d0 >> 16) | ((tensor_d1 & 0xffffu) << 16));
    g1[3] = (int)((tensor_d1 >> 16) | (tile_d0 << 16));   // tile_dim0 @ bits127:112
    g1[4] = (int)(tile_d1 & 0xffffu);               // tile_dim1; tile_dim2=0
    g1[5] = (int)stride0;                           // tensor_dim0_stride[31:0]
    g1[6] = 0;                                      // stride0 hi16 / stride1 lo
    g1[7] = 0;
    i32x4 z4 = {0, 0, 0, 0};
    i32x8 z8 = {0, 0, 0, 0, 0, 0, 0, 0};
    __builtin_amdgcn_tensor_load_to_lds(g0, g1, z4, z4, z8, 0);
}

// C/D 16x16 f32 layout: VGPR r -> row r + 8*(lane>=16); col = lane&15.

// ---------------- prep: fp32 [K][N] -> bf16 transposed [N][Kpad] ----------------
__global__ void k_conv_t(const float* __restrict__ w, bf16_t* __restrict__ out,
                         int K, int N, int Kpad) {
    int idx = blockIdx.x * blockDim.x + threadIdx.x;
    if (idx >= N * Kpad) return;
    int n = idx / Kpad, k = idx - n * Kpad;
    float v = (k < K) ? w[k * N + n] : 0.0f;
    out[idx] = (bf16_t)v;
}

// ---------------- xf = x @ cf_lin1  [NCONF,256]x[256,128] ----------------
__global__ __launch_bounds__(256) void k_xf(const float* __restrict__ x,
                                            const bf16_t* __restrict__ bt,
                                            float* __restrict__ xf) {
    __shared__ alignas(16) bf16_t As[16 * HIDDEN];
    int row0 = blockIdx.x * 16;
    for (int i = threadIdx.x * 4; i < 16 * HIDDEN; i += 256 * 4) {
        v4f v = *(const v4f*)(x + row0 * HIDDEN + i);
#pragma unroll
        for (int j = 0; j < 4; ++j) As[i + j] = (bf16_t)v[j];
    }
    __syncthreads();
    int w = threadIdx.x >> 5, lane = threadIdx.x & 31;
    int col = lane & 15, hh = lane >> 4;
    v8f acc = {};
#pragma unroll
    for (int k0 = 0; k0 < HIDDEN; k0 += 32) {
        v16bf a = load_frag(As, HIDDEN, k0);
        v16bf b = load_frag(bt + (w * 16) * HIDDEN, HIDDEN, k0);
        acc = wmma32(a, b, acc);
    }
#pragma unroll
    for (int r = 0; r < 8; ++r) {
        int row = r + hh * 8;
        xf[(row0 + row) * NF + w * 16 + col] = acc[r];
    }
}

// ---------------- fused edge filter MLP + cosine envelope + scatter-add -------
// edge_attr tile staged by the Tensor Data Mover (TENSORcnt), then converted
// fp32->bf16 in LDS for the WMMA filter MLP.
__global__ __launch_bounds__(256) void k_edge(const float* __restrict__ eattr,
                                              const float* __restrict__ ew,
                                              const int* __restrict__ eidx,
                                              const bf16_t* __restrict__ w1t,
                                              const bf16_t* __restrict__ w2t,
                                              const float* __restrict__ b1,
                                              const float* __restrict__ b2,
                                              const float* __restrict__ xf,
                                              float* __restrict__ agg) {
    __shared__ alignas(16) float  Stage[16 * NG];   // raw fp32 tile from TDM
    __shared__ alignas(16) bf16_t Aat[16 * NGP];
    __shared__ alignas(16) bf16_t Hid[16 * NF];
    __shared__ float cenv[16];
    __shared__ int   srcs[16], dsts[16];
    int e0 = blockIdx.x * 16;

    if (threadIdx.x < 32) {   // wave 0 issues the TDM descriptor (EXEC ignored)
        tdm_load_tile_f32(eattr + (size_t)e0 * NG, Stage,
                          /*tensor_d0=*/NG, /*tensor_d1=*/ECONF,
                          /*tile_d0=*/NG, /*tile_d1=*/16, /*stride0=*/NG);
        if (threadIdx.x < 16) {
            int e = e0 + threadIdx.x;
            srcs[threadIdx.x] = eidx[e];
            dsts[threadIdx.x] = eidx[ECONF + e];
            cenv[threadIdx.x] = 0.5f * (cosf(ew[e] * (PI_F / CUTOFF)) + 1.0f);
        }
        __builtin_amdgcn_s_wait_tensorcnt(0);
    }
    __syncthreads();
    // fp32 -> bf16 with K padding 50 -> 64
    for (int i = threadIdx.x; i < 16 * NGP; i += 256) {
        int er = i >> 6, k = i & 63;
        Aat[i] = (bf16_t)((k < NG) ? Stage[er * NG + k] : 0.0f);
    }
    __syncthreads();

    int w = threadIdx.x >> 5, lane = threadIdx.x & 31;
    int col = lane & 15, hh = lane >> 4;
    // hidden = relu(attr @ w1 + b1)
    v8f acc = {};
#pragma unroll
    for (int k0 = 0; k0 < NGP; k0 += 32) {
        v16bf a = load_frag(Aat, NGP, k0);
        v16bf b = load_frag(w1t + (w * 16) * NGP, NGP, k0);
        acc = wmma32(a, b, acc);
    }
    float bias1 = b1[w * 16 + col];
#pragma unroll
    for (int r = 0; r < 8; ++r) {
        int row = r + hh * 8;
        Hid[row * NF + w * 16 + col] = (bf16_t)fmaxf(acc[r] + bias1, 0.0f);
    }
    __syncthreads();
    // W = (hidden @ w2 + b2) * cenv ; scatter xf[src]*W into agg[dst]
    v8f acc2 = {};
#pragma unroll
    for (int k0 = 0; k0 < NF; k0 += 32) {
        v16bf a = load_frag(Hid, NF, k0);
        v16bf b = load_frag(w2t + (w * 16) * NF, NF, k0);
        acc2 = wmma32(a, b, acc2);
    }
    float bias2 = b2[w * 16 + col];
#pragma unroll
    for (int r = 0; r < 8; ++r) {
        int row = r + hh * 8;
        float wv  = (acc2[r] + bias2) * cenv[row];
        float val = wv * xf[srcs[row] * NF + w * 16 + col];
        atomicAdd(&agg[dsts[row] * NF + w * 16 + col], val);
    }
}

// ------------- h = relu(agg@cf_lin2+b) @ lin_w + lin_b  -> d_out -------------
// agg tile (16x128 fp32, contiguous) staged via TDM, converted to bf16.
__global__ __launch_bounds__(256) void k_h(const float* __restrict__ agg,
                                           const bf16_t* __restrict__ lin2t,
                                           const float* __restrict__ lin2b,
                                           const bf16_t* __restrict__ linwt,
                                           const float* __restrict__ linb,
                                           float* __restrict__ out) {
    __shared__ alignas(16) float  Stage[16 * NF];
    __shared__ alignas(16) bf16_t Aag[16 * NF];
    __shared__ alignas(16) bf16_t T1[16 * HIDDEN];
    int row0 = blockIdx.x * 16;

    if (threadIdx.x < 32) {
        tdm_load_tile_f32(agg + (size_t)row0 * NF, Stage,
                          /*tensor_d0=*/NF, /*tensor_d1=*/NCONF,
                          /*tile_d0=*/NF, /*tile_d1=*/16, /*stride0=*/NF);
        __builtin_amdgcn_s_wait_tensorcnt(0);
    }
    __syncthreads();
    for (int i = threadIdx.x; i < 16 * NF; i += 256)
        Aag[i] = (bf16_t)Stage[i];
    __syncthreads();

    int w = threadIdx.x >> 5, lane = threadIdx.x & 31;
    int col = lane & 15, hh = lane >> 4;
#pragma unroll
    for (int t = 0; t < 2; ++t) {
        int nt = w + t * 8;
        v8f acc = {};
#pragma unroll
        for (int k0 = 0; k0 < NF; k0 += 32) {
            v16bf a = load_frag(Aag, NF, k0);
            v16bf b = load_frag(lin2t + (nt * 16) * NF, NF, k0);
            acc = wmma32(a, b, acc);
        }
        float bias = lin2b[nt * 16 + col];
#pragma unroll
        for (int r = 0; r < 8; ++r) {
            int row = r + hh * 8;
            T1[row * HIDDEN + nt * 16 + col] = (bf16_t)fmaxf(acc[r] + bias, 0.0f);
        }
    }
    __syncthreads();
#pragma unroll
    for (int t = 0; t < 2; ++t) {
        int nt = w + t * 8;
        v8f acc = {};
#pragma unroll
        for (int k0 = 0; k0 < HIDDEN; k0 += 32) {
            v16bf a = load_frag(T1, HIDDEN, k0);
            v16bf b = load_frag(linwt + (nt * 16) * HIDDEN, HIDDEN, k0);
            acc = wmma32(a, b, acc);
        }
        float bias = linb[nt * 16 + col];
#pragma unroll
        for (int r = 0; r < 8; ++r) {
            int row = r + hh * 8;
            out[(row0 + row) * HIDDEN + nt * 16 + col] = acc[r] + bias;
        }
    }
}

// ---------------- GIN: segment max (each graph node owns 10 rows) ------------
__global__ void k_segmax(const float* __restrict__ x, float* __restrict__ xagg) {
    int g = blockIdx.x, c = threadIdx.x;
    float m = -3.402823e38f;
#pragma unroll
    for (int j = 0; j < PER_G; ++j)
        m = fmaxf(m, x[(g * PER_G + j) * HIDDEN + c]);
    xagg[g * HIDDEN + c] = m;
}

// ---------------- GIN edge: relu(x_agg[sg]+edge_emb) scatter-add -------------
__global__ void k_ginedge(const int* __restrict__ eidx, const int* __restrict__ eatt,
                          const float* __restrict__ bemb, const float* __restrict__ xagg,
                          float* __restrict__ aggg) {
    int e = blockIdx.x, c = threadIdx.x;
    int a0 = eatt[e * 3 + 0], a1 = eatt[e * 3 + 1], a2 = eatt[e * 3 + 2];
    int sg = eidx[e], dg = eidx[EGRAPH + e];
    float emb = bemb[(0 * 5 + a0) * HIDDEN + c]
              + bemb[(1 * 5 + a1) * HIDDEN + c]
              + bemb[(2 * 5 + a2) * HIDDEN + c];
    float v = fmaxf(xagg[sg * HIDDEN + c] + emb, 0.0f);
    atomicAdd(&aggg[dg * HIDDEN + c], v);
}

// -------- GIN GEMM (mode 0: A=(1+eps)*xagg+aggg ; mode 1: A=relu(bn(u))) -----
__global__ __launch_bounds__(256) void k_gingemm(const float* __restrict__ A0,
                                                 const float* __restrict__ A1,
                                                 const float* __restrict__ scale,
                                                 const float* __restrict__ shift,
                                                 const float* __restrict__ epsp,
                                                 int mode,
                                                 const bf16_t* __restrict__ Bt,
                                                 const float* __restrict__ bias,
                                                 float* __restrict__ out) {
    __shared__ alignas(16) bf16_t As[16 * HIDDEN];
    int row0 = blockIdx.x * 16;
    if (mode == 0) {
        float e1 = 1.0f + epsp[0];
        for (int i = threadIdx.x; i < 16 * HIDDEN; i += 256)
            As[i] = (bf16_t)(e1 * A0[row0 * HIDDEN + i] + A1[row0 * HIDDEN + i]);
    } else {
        for (int i = threadIdx.x; i < 16 * HIDDEN; i += 256) {
            int c = i & (HIDDEN - 1);
            As[i] = (bf16_t)fmaxf(A0[row0 * HIDDEN + i] * scale[c] + shift[c], 0.0f);
        }
    }
    __syncthreads();
    int w = threadIdx.x >> 5, lane = threadIdx.x & 31;
    int col = lane & 15, hh = lane >> 4;
#pragma unroll
    for (int t = 0; t < 2; ++t) {
        int nt = w + t * 8;
        v8f acc = {};
#pragma unroll
        for (int k0 = 0; k0 < HIDDEN; k0 += 32) {
            v16bf a = load_frag(As, HIDDEN, k0);
            v16bf b = load_frag(Bt + (nt * 16) * HIDDEN, HIDDEN, k0);
            acc = wmma32(a, b, acc);
        }
        float bs = bias[nt * 16 + col];
#pragma unroll
        for (int r = 0; r < 8; ++r) {
            int row = r + hh * 8;
            out[(row0 + row) * HIDDEN + nt * 16 + col] = acc[r] + bs;
        }
    }
}

// ---------------- BatchNorm stats (two-pass, biased variance) ----------------
__global__ void k_bnstats(const float* __restrict__ u, float* __restrict__ sum,
                          float* __restrict__ sumsq) {
    int c = threadIdx.x;
    float s = 0.0f, q = 0.0f;
    for (int r = blockIdx.x; r < NGRAPH; r += gridDim.x) {
        float v = u[r * HIDDEN + c];
        s += v; q += v * v;
    }
    atomicAdd(&sum[c], s);
    atomicAdd(&sumsq[c], q);
}

__global__ void k_bnfinal(const float* __restrict__ sum, const float* __restrict__ sumsq,
                          const float* __restrict__ g, const float* __restrict__ b,
                          float* __restrict__ scale, float* __restrict__ shift) {
    int c = threadIdx.x;
    float mu  = sum[c] * (1.0f / NGRAPH);
    float var = sumsq[c] * (1.0f / NGRAPH) - mu * mu;
    float sc  = g[c] * rsqrtf(var + 1e-5f);
    scale[c] = sc;
    shift[c] = b[c] - mu * sc;
}

// ---------------- out = x + h + bn2(v)[batch]  (h already in d_out) ----------
__global__ void k_final(const float* __restrict__ x, const int* __restrict__ batch,
                        const float* __restrict__ v, const float* __restrict__ scale2,
                        const float* __restrict__ shift2, float* __restrict__ out) {
    int i = blockIdx.x, c = threadIdx.x;
    int g = batch[i];
    out[i * HIDDEN + c] += x[i * HIDDEN + c]
                         + v[g * HIDDEN + c] * scale2[c] + shift2[c];
}

// =============================== launcher ====================================
extern "C" void kernel_launch(void* const* d_in, const int* in_sizes, int n_in,
                              void* d_out, int out_size, void* d_ws, size_t ws_size,
                              hipStream_t stream) {
    (void)in_sizes; (void)n_in; (void)out_size; (void)ws_size;
    // -------- inputs (setup_inputs order) --------
    const float* x        = (const float*)d_in[0];
    const int*   batch    = (const int*)  d_in[1];
    const int*   ei_conf  = (const int*)  d_in[2];
    const float* ew_conf  = (const float*)d_in[3];
    const float* ea_conf  = (const float*)d_in[4];
    const int*   ei_graph = (const int*)  d_in[5];
    const int*   ea_graph = (const int*)  d_in[6];
    const float* mlp_w1   = (const float*)d_in[7];
    const float* mlp_b1   = (const float*)d_in[8];
    const float* mlp_w2   = (const float*)d_in[9];
    const float* mlp_b2   = (const float*)d_in[10];
    const float* cf_lin1  = (const float*)d_in[11];
    const float* cf_lin2  = (const float*)d_in[12];
    const float* cf_l2b   = (const float*)d_in[13];
    const float* lin_w    = (const float*)d_in[14];
    const float* lin_b    = (const float*)d_in[15];
    const float* bond_emb = (const float*)d_in[16];
    const float* gin_eps  = (const float*)d_in[17];
    const float* gin_w1   = (const float*)d_in[18];
    const float* gin_b1   = (const float*)d_in[19];
    const float* bn1_g    = (const float*)d_in[20];
    const float* bn1_b    = (const float*)d_in[21];
    const float* gin_w2   = (const float*)d_in[22];
    const float* gin_b2   = (const float*)d_in[23];
    const float* bn2_g    = (const float*)d_in[24];
    const float* bn2_b    = (const float*)d_in[25];
    float* out = (float*)d_out;

    // -------- workspace layout --------
    char* ws = (char*)d_ws;
    size_t off = 0;
    auto take = [&](size_t bytes) { size_t r = off; off += (bytes + 255) & ~(size_t)255; return r; };
    float*  xf      = (float*)(ws + take((size_t)NCONF * NF * 4));
    float*  agg     = (float*)(ws + take((size_t)NCONF * NF * 4));
    float*  xagg    = (float*)(ws + take((size_t)NGRAPH * HIDDEN * 4));
    float*  aggg    = (float*)(ws + take((size_t)NGRAPH * HIDDEN * 4));
    float*  u       = (float*)(ws + take((size_t)NGRAPH * HIDDEN * 4));
    float*  v       = (float*)(ws + take((size_t)NGRAPH * HIDDEN * 4));
    float*  sum1    = (float*)(ws + take(HIDDEN * 4));
    float*  sq1     = (float*)(ws + take(HIDDEN * 4));
    float*  sum2    = (float*)(ws + take(HIDDEN * 4));
    float*  sq2     = (float*)(ws + take(HIDDEN * 4));
    float*  scale1  = (float*)(ws + take(HIDDEN * 4));
    float*  shift1  = (float*)(ws + take(HIDDEN * 4));
    float*  scale2  = (float*)(ws + take(HIDDEN * 4));
    float*  shift2  = (float*)(ws + take(HIDDEN * 4));
    bf16_t* lin1t   = (bf16_t*)(ws + take((size_t)NF * HIDDEN * 2));     // [128][256]
    bf16_t* w1t     = (bf16_t*)(ws + take((size_t)NF * NGP * 2));        // [128][64]
    bf16_t* w2t     = (bf16_t*)(ws + take((size_t)NF * NF * 2));         // [128][128]
    bf16_t* lin2t   = (bf16_t*)(ws + take((size_t)HIDDEN * NF * 2));     // [256][128]
    bf16_t* linwt   = (bf16_t*)(ws + take((size_t)HIDDEN * HIDDEN * 2)); // [256][256]
    bf16_t* gw1t    = (bf16_t*)(ws + take((size_t)HIDDEN * HIDDEN * 2));
    bf16_t* gw2t    = (bf16_t*)(ws + take((size_t)HIDDEN * HIDDEN * 2));

    // -------- zero accumulators / stats (graph-capturable async memsets) -----
    (void)hipMemsetAsync(agg,  0, (size_t)NCONF * NF * 4,      stream);
    (void)hipMemsetAsync(aggg, 0, (size_t)NGRAPH * HIDDEN * 4, stream);
    (void)hipMemsetAsync(sum1, 0, HIDDEN * 4, stream);
    (void)hipMemsetAsync(sq1,  0, HIDDEN * 4, stream);
    (void)hipMemsetAsync(sum2, 0, HIDDEN * 4, stream);
    (void)hipMemsetAsync(sq2,  0, HIDDEN * 4, stream);

    // -------- weight prep: bf16, transposed, K padded --------
    auto prep = [&](const float* w, bf16_t* o, int K, int N, int Kpad) {
        int total = N * Kpad;
        k_conv_t<<<(total + 255) / 256, 256, 0, stream>>>(w, o, K, N, Kpad);
    };
    prep(cf_lin1, lin1t, HIDDEN, NF, HIDDEN);
    prep(mlp_w1,  w1t,   NG,     NF, NGP);
    prep(mlp_w2,  w2t,   NF,     NF, NF);
    prep(cf_lin2, lin2t, NF,     HIDDEN, NF);
    prep(lin_w,   linwt, HIDDEN, HIDDEN, HIDDEN);
    prep(gin_w1,  gw1t,  HIDDEN, HIDDEN, HIDDEN);
    prep(gin_w2,  gw2t,  HIDDEN, HIDDEN, HIDDEN);

    // -------- CFConv branch --------
    k_xf  <<<NCONF / 16, 256, 0, stream>>>(x, lin1t, xf);
    k_edge<<<ECONF / 16, 256, 0, stream>>>(ea_conf, ew_conf, ei_conf, w1t, w2t,
                                           mlp_b1, mlp_b2, xf, agg);
    k_h   <<<NCONF / 16, 256, 0, stream>>>(agg, lin2t, cf_l2b, linwt, lin_b, out);

    // -------- GIN branch --------
    k_segmax <<<NGRAPH, HIDDEN, 0, stream>>>(x, xagg);
    k_ginedge<<<EGRAPH, HIDDEN, 0, stream>>>(ei_graph, ea_graph, bond_emb, xagg, aggg);
    k_gingemm<<<NGRAPH / 16, 256, 0, stream>>>(xagg, aggg, nullptr, nullptr, gin_eps,
                                               0, gw1t, gin_b1, u);
    k_bnstats<<<128, HIDDEN, 0, stream>>>(u, sum1, sq1);
    k_bnfinal<<<1, HIDDEN, 0, stream>>>(sum1, sq1, bn1_g, bn1_b, scale1, shift1);
    k_gingemm<<<NGRAPH / 16, 256, 0, stream>>>(u, nullptr, scale1, shift1, gin_eps,
                                               1, gw2t, gin_b2, v);
    k_bnstats<<<128, HIDDEN, 0, stream>>>(v, sum2, sq2);
    k_bnfinal<<<1, HIDDEN, 0, stream>>>(sum2, sq2, bn2_g, bn2_b, scale2, shift2);

    // -------- residual combine --------
    k_final<<<NCONF, HIDDEN, 0, stream>>>(x, batch, v, scale2, shift2, out);
}
